// GRU_3410204033806
// MI455X (gfx1250) — compile-verified, matrix operand order checked by reading, not run on hardware
//
#include <hip/hip_runtime.h>
#include <hip/hip_bf16.h>
#include <math.h>

// ---------------------------------------------------------------------------
// GRU (2-layer) + top-k selection + waterfall rebalance for MI455X (gfx1250)
//
//  K0: f32 -> f16 pre-conversion of Wih (removes cvt from GEMM hot loop)
//  K1: GI = X(64512x2000) @ Wih0^T + bih0   -> f16 WMMA GEMM, 32x384 block tile
//  K2: layer-0 scan (Whh0 in LDS as f16, 12 WMMA/step), writes H0[B,T,H]
//  K3: GI = H0(64512x128) @ Wih1^T + bih1   -> same GEMM, K=128
//  K4: layer-1 scan, writes only final h (B,H)
//  K5: per-row head: scores matvec, iterative top-50 (jax tie-break),
//      softmax(silu(.)), scatter, 50-iter rebalance on wave0 via shfl_xor.
//
// Workspace: GI 99.1MB + H0 33.0MB + hT 128KB + W16 1.5MB (~133.8 MB of d_ws)
// ---------------------------------------------------------------------------

typedef __attribute__((ext_vector_type(16))) _Float16 v16h;
typedef __attribute__((ext_vector_type(8)))  float    v8f;

#define CB   256      // batch
#define CT   252      // timesteps
#define CN   2000     // features
#define CH   128      // hidden
#define C3H  384
#define CM   (CB*CT)  // 64512 rows for the big GEMMs
#define NSEL 50

union V16H { v16h v; float4 f4[2]; };

static __device__ __forceinline__ float sigmoidf_(float x) {
  return 1.0f / (1.0f + __expf(-x));
}
static __device__ __forceinline__ float waveSum(float x) {
  #pragma unroll
  for (int o = 16; o > 0; o >>= 1) x += __shfl_xor(x, o, 32);
  return x;
}
static __device__ __forceinline__ int waveOr(int x) {
  #pragma unroll
  for (int o = 16; o > 0; o >>= 1) x |= __shfl_xor(x, o, 32);
  return x;
}

// ---------------------------------------------------------------------------
// K0: bulk f32 -> f16 conversion, 8 elems/thread (n % 8 == 0)
// ---------------------------------------------------------------------------
__global__ __launch_bounds__(256) void cvt_f16_kernel(
    const float* __restrict__ in, _Float16* __restrict__ out, int n)
{
  const int i = (blockIdx.x * 256 + threadIdx.x) * 8;
  if (i < n) {
    const float4 a = *reinterpret_cast<const float4*>(in + i);
    const float4 b = *reinterpret_cast<const float4*>(in + i + 4);
    _Float16 o[8] = {(_Float16)a.x, (_Float16)a.y, (_Float16)a.z, (_Float16)a.w,
                     (_Float16)b.x, (_Float16)b.y, (_Float16)b.z, (_Float16)b.w};
    *reinterpret_cast<float4*>(out + i) = *reinterpret_cast<float4*>(o);
  }
}

// ---------------------------------------------------------------------------
// K1/K3: C[M,384] = A[M,K] @ W16[384,K]^T + bias   (A f32, W pre-cvt f16)
// grid = M/32; block = 256 (8 waves). Block tile: 32 rows x 384 cols.
// Wave w owns cols [48w, 48w+48): 2 M-tiles x 3 N-tiles = 6 wmma / k-step.
// A strip (32 x 32) staged once per k-step in LDS, shared by all waves.
// Main loop unguarded; K-remainder handled in a guarded tail iteration.
// ---------------------------------------------------------------------------
__global__ __launch_bounds__(256) void wmma_gemm384_kernel(
    const float* __restrict__ A, const _Float16* __restrict__ W16,
    const float* __restrict__ bias, float* __restrict__ C, int K)
{
  __shared__ _Float16 sA[32 * 32];   // 2 KB

  const int tid  = threadIdx.x;
  const int wv   = tid >> 5;
  const int lane = tid & 31;
  const int l15  = lane & 15;
  const int hi   = lane >> 4;       // 0 or 1
  const int hiK  = hi * 8;

  const int rowBase = blockIdx.x * 32;
  const int colBase = wv * 48;

  // staging coords: 32 rows x 32 k, 4 contiguous floats per thread
  const int sr = tid >> 3;          // 0..31
  const int sc = (tid & 7) * 4;     // 0..28
  const float* __restrict__ aRow = A + (size_t)(rowBase + sr) * K;

  v8f acc[2][3];
  #pragma unroll
  for (int mt = 0; mt < 2; ++mt)
    #pragma unroll
    for (int nt = 0; nt < 3; ++nt)
      #pragma unroll
      for (int r = 0; r < 8; ++r) acc[mt][nt][r] = 0.0f;

  const int nFull = K / 32;
  const bool tail = (K % 32) != 0;

  for (int it = 0; it < nFull; ++it) {
    const int kb = it * 32;
    // --- stage A strip (unguarded, float4 load + cvt, 8B LDS store) ---
    {
      const float4 v = *reinterpret_cast<const float4*>(aRow + kb + sc);
      _Float16 o[4] = {(_Float16)v.x, (_Float16)v.y, (_Float16)v.z, (_Float16)v.w};
      *reinterpret_cast<float2*>(&sA[sr * 32 + sc]) = *reinterpret_cast<float2*>(o);
    }
    __syncthreads();

    // --- A fragments (two 16B LDS runs per lane per M-tile) ---
    V16H af[2];
    #pragma unroll
    for (int mt = 0; mt < 2; ++mt) {
      const int m = mt * 16 + l15;
      af[mt].f4[0] = *reinterpret_cast<const float4*>(&sA[m * 32 + hiK]);
      af[mt].f4[1] = *reinterpret_cast<const float4*>(&sA[m * 32 + 16 + hiK]);
    }
    // --- B fragments straight from f16 weights (two 16B global loads) ---
    #pragma unroll
    for (int nt = 0; nt < 3; ++nt) {
      const int col = colBase + nt * 16 + l15;
      const _Float16* wr = W16 + (size_t)col * K;
      V16H bf;
      bf.f4[0] = *reinterpret_cast<const float4*>(wr + kb + hiK);
      bf.f4[1] = *reinterpret_cast<const float4*>(wr + kb + 16 + hiK);
      #pragma unroll
      for (int mt = 0; mt < 2; ++mt)
        acc[mt][nt] = __builtin_amdgcn_wmma_f32_16x16x32_f16(
            false, af[mt].v, false, bf.v, (short)0, acc[mt][nt], false, false);
    }
    __syncthreads();
  }

  if (tail) {
    const int kb = nFull * 32;
    {
      float4 v = make_float4(0, 0, 0, 0);
      if (kb + sc < K)                 // K % 4 == 0 -> whole float4 in range
        v = *reinterpret_cast<const float4*>(aRow + kb + sc);
      _Float16 o[4] = {(_Float16)v.x, (_Float16)v.y, (_Float16)v.z, (_Float16)v.w};
      *reinterpret_cast<float2*>(&sA[sr * 32 + sc]) = *reinterpret_cast<float2*>(o);
    }
    __syncthreads();

    V16H af[2];
    #pragma unroll
    for (int mt = 0; mt < 2; ++mt) {
      const int m = mt * 16 + l15;
      af[mt].f4[0] = *reinterpret_cast<const float4*>(&sA[m * 32 + hiK]);
      af[mt].f4[1] = *reinterpret_cast<const float4*>(&sA[m * 32 + 16 + hiK]);
    }
    const int k1 = kb + hiK;
    const int k2 = kb + 16 + hiK;
    #pragma unroll
    for (int nt = 0; nt < 3; ++nt) {
      const int col = colBase + nt * 16 + l15;
      const _Float16* wr = W16 + (size_t)col * K;
      V16H bf;
      bf.f4[0] = make_float4(0, 0, 0, 0);
      bf.f4[1] = make_float4(0, 0, 0, 0);
      if (k1 < K) bf.f4[0] = *reinterpret_cast<const float4*>(wr + k1);  // K%8==0
      if (k2 < K) bf.f4[1] = *reinterpret_cast<const float4*>(wr + k2);
      #pragma unroll
      for (int mt = 0; mt < 2; ++mt)
        acc[mt][nt] = __builtin_amdgcn_wmma_f32_16x16x32_f16(
            false, af[mt].v, false, bf.v, (short)0, acc[mt][nt], false, false);
    }
  }

  // epilogue: D layout -> row = rowBase + 16*mt + r + 8*hi, col per N-tile
  #pragma unroll
  for (int nt = 0; nt < 3; ++nt) {
    const int col  = colBase + nt * 16 + l15;
    const float bc = bias[col];
    #pragma unroll
    for (int mt = 0; mt < 2; ++mt) {
      #pragma unroll
      for (int r = 0; r < 8; ++r) {
        const int row = rowBase + mt * 16 + r + 8 * hi;
        C[(size_t)row * C3H + col] = acc[mt][nt][r] + bc;
      }
    }
  }
}

// ---------------------------------------------------------------------------
// K2/K4: GRU recurrent scan. grid = 16 blocks x 256 thr; block owns 16 batch
// rows. Whh (384x128) resident in LDS as f16; B-fragments hoisted to regs.
// Per step: 12 WMMA (3 gates x 4 k-steps), gate math, h update.
//   outAll  != nullptr : write h for every t (layout [B,T,H])
//   outFinal!= nullptr : write final h (layout [B,H])
// dyn LDS = 384*128*2 (Whh f16) + 16*128*4 (h f32) + 16*128*2 (h f16) = 110592B
// ---------------------------------------------------------------------------
__global__ __launch_bounds__(256) void gru_scan_kernel(
    const float* __restrict__ GI, const float* __restrict__ Whh,
    const float* __restrict__ bhh, float* __restrict__ outAll,
    float* __restrict__ outFinal)
{
  extern __shared__ unsigned char smem[];
  _Float16* sW   = (_Float16*)smem;                                   // [384][128]
  float*    sh   = (float*)   (smem + C3H * CH * 2);                  // [16][128]
  _Float16* sh16 = (_Float16*)(smem + C3H * CH * 2 + 16 * CH * 4);    // [16][128]

  const int tid  = threadIdx.x;
  const int wv   = tid >> 5;
  const int lane = tid & 31;
  const int l15  = lane & 15;
  const int hi   = lane >> 4;
  const int hiK  = hi * 8;
  const int rowBase = blockIdx.x * 16;
  const int hcol = wv * 16 + l15;            // this lane's hidden column

  for (int i = tid; i < C3H * CH; i += 256) sW[i] = (_Float16)Whh[i];
  for (int i = tid; i < 16 * CH; i += 256) { sh[i] = 0.0f; sh16[i] = (_Float16)0.0f; }
  __syncthreads();

  // loop-invariant B fragments: B[k][n] = Whh[n][k], rows contiguous in k
  v16h bfr[3][4];
  #pragma unroll
  for (int g = 0; g < 3; ++g) {
    const int n = g * CH + hcol;
    #pragma unroll
    for (int i = 0; i < 4; ++i) {
      V16H u;
      u.f4[0] = *reinterpret_cast<const float4*>(&sW[n * CH + i * 32 + hiK]);
      u.f4[1] = *reinterpret_cast<const float4*>(&sW[n * CH + i * 32 + 16 + hiK]);
      bfr[g][i] = u.v;
    }
  }
  const float bR = bhh[hcol], bZ = bhh[CH + hcol], bN = bhh[2 * CH + hcol];

  for (int t = 0; t < CT; ++t) {
    v8f aR, aZ, aN;
    #pragma unroll
    for (int r = 0; r < 8; ++r) { aR[r] = 0.0f; aZ[r] = 0.0f; aN[r] = 0.0f; }

    #pragma unroll
    for (int i = 0; i < 4; ++i) {
      V16H a;
      a.f4[0] = *reinterpret_cast<const float4*>(&sh16[l15 * CH + i * 32 + hiK]);
      a.f4[1] = *reinterpret_cast<const float4*>(&sh16[l15 * CH + i * 32 + 16 + hiK]);
      aR = __builtin_amdgcn_wmma_f32_16x16x32_f16(false, a.v, false, bfr[0][i], (short)0, aR, false, false);
      aZ = __builtin_amdgcn_wmma_f32_16x16x32_f16(false, a.v, false, bfr[1][i], (short)0, aZ, false, false);
      aN = __builtin_amdgcn_wmma_f32_16x16x32_f16(false, a.v, false, bfr[2][i], (short)0, aN, false, false);
    }
    __syncthreads();   // all sh16 reads complete before overwrite

    #pragma unroll
    for (int r = 0; r < 8; ++r) {
      const int m   = r + 8 * hi;
      const int row = rowBase + m;                       // batch index
      const float* gi = GI + ((size_t)row * CT + t) * C3H;
      const float ir  = gi[hcol];
      const float iz  = gi[CH + hcol];
      const float inn = gi[2 * CH + hcol];
      const float hr = aR[r] + bR;
      const float hz = aZ[r] + bZ;
      const float hn = aN[r] + bN;
      const float rg = sigmoidf_(ir + hr);
      const float zg = sigmoidf_(iz + hz);
      const float ng = tanhf(inn + rg * hn);
      const float ho = sh[m * CH + hcol];
      const float hv = (1.0f - zg) * ng + zg * ho;
      sh[m * CH + hcol]   = hv;
      sh16[m * CH + hcol] = (_Float16)hv;
      if (outAll) outAll[((size_t)row * CT + t) * CH + hcol] = hv;
    }
    __syncthreads();
  }

  if (outFinal) {
    #pragma unroll
    for (int r = 0; r < 8; ++r) {
      const int m = r + 8 * hi;
      outFinal[(size_t)(rowBase + m) * CH + hcol] = sh[m * CH + hcol];
    }
  }
}

// ---------------------------------------------------------------------------
// K5: per-batch-row head. block(256) per row.
// ---------------------------------------------------------------------------
__global__ __launch_bounds__(256) void head_kernel(
    const float* __restrict__ hT, const float* __restrict__ Wsel,
    const float* __restrict__ bsel, const float* __restrict__ Wwt,
    const float* __restrict__ bwt, float* __restrict__ out)
{
  __shared__ float shh[CH];
  __shared__ float sc[CN];
  __shared__ float rv[256];
  __shared__ int   ri[256];
  __shared__ int   selIdx[NSEL];
  __shared__ float sw[NSEL];

  const int tid = threadIdx.x;
  const int b   = blockIdx.x;

  if (tid < CH) shh[tid] = hT[(size_t)b * CH + tid];
  __syncthreads();

  // scores (sigmoid is monotone -> ranking identical without it)
  for (int j = tid; j < CN; j += 256) {
    const float* wr = Wsel + (size_t)j * CH;
    float s = bsel[j];
    for (int k = 0; k < CH; ++k) s += shh[k] * wr[k];
    sc[j] = s;
  }
  __syncthreads();

  // iterative top-50, jax tie-break (equal value -> lower index first)
  for (int it = 0; it < NSEL; ++it) {
    float bv = -INFINITY; int bi = 0x7fffffff;
    for (int j = tid; j < CN; j += 256) {
      const float v = sc[j];
      if (v > bv || (v == bv && j < bi)) { bv = v; bi = j; }
    }
    rv[tid] = bv; ri[tid] = bi;
    __syncthreads();
    for (int s = 128; s > 0; s >>= 1) {
      if (tid < s) {
        const float v2 = rv[tid + s]; const int i2 = ri[tid + s];
        if (v2 > rv[tid] || (v2 == rv[tid] && i2 < ri[tid])) { rv[tid] = v2; ri[tid] = i2; }
      }
      __syncthreads();
    }
    if (tid == 0) { selIdx[it] = ri[0]; sc[ri[0]] = -INFINITY; }
    __syncthreads();
  }

  // w = softmax(silu(h @ Wwt^T + bwt))
  if (tid < NSEL) {
    const float* wr = Wwt + (size_t)tid * CH;
    float s = bwt[tid];
    for (int k = 0; k < CH; ++k) s += shh[k] * wr[k];
    sw[tid] = s * sigmoidf_(s);          // silu
  }
  __syncthreads();
  if (tid == 0) {
    float mx = -INFINITY;
    for (int i = 0; i < NSEL; ++i) mx = fmaxf(mx, sw[i]);
    float sum = 0.0f;
    for (int i = 0; i < NSEL; ++i) { const float e = __expf(sw[i] - mx); sw[i] = e; sum += e; }
    for (int i = 0; i < NSEL; ++i) sw[i] /= sum;
  }
  __syncthreads();

  // zero the output row, then barrier before scatter
  for (int j = tid; j < CN; j += 256) out[(size_t)b * CN + j] = 0.0f;
  __syncthreads();

  // waterfall rebalance on wave 0 (lane e and e+32; non-selected cols stay 0)
  if (tid < 32) {
    const float UB = 0.2f, LB = 0.0f;
    const int  e0 = tid, e1 = tid + 32;
    const bool v1 = (e1 < NSEL);
    float old0 = sw[e0];
    float old1 = v1 ? sw[e1] : 0.0f;
    float wc0 = fminf(fmaxf(old0, LB), UB);
    float wc1 = v1 ? fminf(fmaxf(old1, LB), UB) : 0.0f;
    bool active = true;
    for (int it = 0; it < NSEL; ++it) {
      const float leftover = waveSum((old0 - wc0) + (old1 - wc1));
      const bool nom0 = (wc0 != UB);                 // sel true for valid entries
      const bool nom1 = v1 && (wc1 != UB);
      const float nsum = waveSum((nom0 ? wc0 : 0.0f) + (nom1 ? wc1 : 0.0f));
      const int  has  = waveOr((nom0 || nom1) ? 1 : 0);
      const float den = (nsum == 0.0f) ? 1.0f : nsum;
      const float w20 = wc0 + (nom0 ? leftover * wc0 / den : 0.0f);
      const float w21 = wc1 + (nom1 ? leftover * wc1 / den : 0.0f);
      const int over = waveOr(((w20 > UB) || (v1 && (w21 > UB))) ? 1 : 0);
      const float w30 = over ? fminf(fmaxf(w20, LB), UB) : w20;
      const float w31 = over ? fminf(fmaxf(w21, LB), UB) : w21;
      const bool doit = active && (has != 0);
      old0 = doit ? w20 : old0;  old1 = doit ? w21 : old1;
      wc0  = doit ? w30 : wc0;   wc1  = doit ? w31 : wc1;
      active = doit && (over != 0);
    }
    out[(size_t)b * CN + selIdx[e0]] = wc0;
    if (v1) out[(size_t)b * CN + selIdx[e1]] = wc1;
  }
}

// ---------------------------------------------------------------------------
extern "C" void kernel_launch(void* const* d_in, const int* in_sizes, int n_in,
                              void* d_out, int out_size, void* d_ws, size_t ws_size,
                              hipStream_t stream) {
  (void)in_sizes; (void)n_in; (void)out_size; (void)ws_size;
  const float* x    = (const float*)d_in[0];
  const float* Wih0 = (const float*)d_in[1];
  const float* Whh0 = (const float*)d_in[2];
  const float* bih0 = (const float*)d_in[3];
  const float* bhh0 = (const float*)d_in[4];
  const float* Wih1 = (const float*)d_in[5];
  const float* Whh1 = (const float*)d_in[6];
  const float* bih1 = (const float*)d_in[7];
  const float* bhh1 = (const float*)d_in[8];
  const float* Wsel = (const float*)d_in[9];
  const float* bsel = (const float*)d_in[10];
  const float* Wwt  = (const float*)d_in[11];
  const float* bwt  = (const float*)d_in[12];
  float* outp = (float*)d_out;

  // workspace carve-up (~133.8 MB)
  float*    GI  = (float*)d_ws;                    // [64512][384], reused L0/L1
  float*    H0  = GI + (size_t)CM * C3H;           // [B][T][H]
  float*    hT  = H0 + (size_t)CM * CH;            // [B][H]
  _Float16* W16 = (_Float16*)(hT + (size_t)CB * CH); // [384][K], reused L0/L1

  const dim3 gGemm(CM / 32);                       // 2016 blocks
  const size_t scanLds = (size_t)C3H * CH * 2 + 16 * CH * 4 + 16 * CH * 2; // 110592B

  // L0 input projection (K = 2000)
  cvt_f16_kernel<<<(C3H * CN) / (256 * 8), 256, 0, stream>>>(Wih0, W16, C3H * CN);
  wmma_gemm384_kernel<<<gGemm, 256, 0, stream>>>(x, W16, bih0, GI, CN);
  // L0 scan -> H0
  gru_scan_kernel<<<CB / 16, 256, scanLds, stream>>>(GI, Whh0, bhh0, H0, nullptr);
  // L1 input projection (K = 128), GI and W16 reused
  cvt_f16_kernel<<<(C3H * CH) / (256 * 8), 256, 0, stream>>>(Wih1, W16, C3H * CH);
  wmma_gemm384_kernel<<<gGemm, 256, 0, stream>>>(H0, W16, bih1, GI, CH);
  // L1 scan -> final hidden only
  gru_scan_kernel<<<CB / 16, 256, scanLds, stream>>>(GI, Whh1, bhh1, nullptr, hT);
  // head: scores, top-50, softmax(silu), scatter + rebalance
  head_kernel<<<CB, 256, 0, stream>>>(hT, Wsel, bsel, Wwt, bwt, outp);
}
